// RepresentationSimilarity_14224931684811
// MI455X (gfx1250) — matrix operand order repaired and checked by reading, not compiled, not run on hardware
//
#include <hip/hip_runtime.h>

// ---------------------------------------------------------------------------
// Cosine-similarity = row-normalized GEMM on gfx1250 (MI455X, wave32, WMMA)
//   out[x,t] = dot(H[x], T[t]) * invnorm(H[x]) * invnorm(T[t])
//
// Compute-bound (AI ~327 FLOP/B). fp32 matrix WMMA (16x16x4) runs ~8x slower
// per FLOP than the bf16 pipe (16x16x32), so we use bf16x3 split precision:
//   x = hi + lo  (hi = bf16 RNE of x, lo = bf16(x - hi))
//   a*b ~= hi_a*hi_b + hi_a*lo_b + lo_a*hi_b          (fp32 accumulate)
// 3x matrix work at ~8x rate => ~2.7x faster than f32 WMMA, ~1e-5 rel error.
//
// This revision software-pipelines the K loop with double-buffered LDS:
// global loads for slab i+1 issue before the WMMAs of slab i, so HBM/L2
// latency hides under matrix work (removes the s_wait_loadcnt 0x0 stall
// observed before the conversion code in the single-buffered version).
// ---------------------------------------------------------------------------

typedef float  v8f    __attribute__((ext_vector_type(8)));
typedef __bf16 v16bf  __attribute__((ext_vector_type(16)));
typedef __bf16 v8bf   __attribute__((ext_vector_type(8)));
typedef __bf16 bf16x4 __attribute__((ext_vector_type(4)));

#define D_DIM 1024   // feature dim (reference-fixed)
#define BM    128    // block tile M
#define BN    128    // block tile N
#define KS    32     // K slab staged in LDS (= one bf16 WMMA K)
#define LDPB  40     // padded LDS row stride (bf16): 80B -> 16B-aligned b128,
                     // banks (20*m)%64 distinct for m=0..15 -> conflict-free

// ---------------------------------------------------------------------------
// Pass 1: per-row inverse norm  inv[row] = 1 / max(||X[row]||, 1e-8)   (fp32)
// ---------------------------------------------------------------------------
__global__ __launch_bounds__(256)
void rs_rownorm_kernel(const float* __restrict__ X, float* __restrict__ inv) {
    __shared__ float red[256];
    const int row = blockIdx.x;
    const int tid = threadIdx.x;
    const float4 v = ((const float4*)(X + (size_t)row * D_DIM))[tid];
    red[tid] = v.x * v.x + v.y * v.y + v.z * v.z + v.w * v.w;
    __syncthreads();
    #pragma unroll
    for (int off = 128; off > 0; off >>= 1) {
        if (tid < off) red[tid] += red[tid + off];
        __syncthreads();
    }
    if (tid == 0) {
        inv[row] = 1.0f / fmaxf(sqrtf(red[0]), 1e-8f);
    }
}

// --- split one float4 into bf16 hi/lo quads (RNE; lo = residual) -----------
__device__ __forceinline__ void rs_split4(const float4 v, bf16x4& hi, bf16x4& lo) {
    const float f[4] = {v.x, v.y, v.z, v.w};
    #pragma unroll
    for (int i = 0; i < 4; ++i) {
        const __bf16 h = (__bf16)f[i];
        hi[i] = h;
        lo[i] = (__bf16)(f[i] - (float)h);
    }
}

// --- fragment loads from LDS (contiguous k = 16h + s order, same for A and B, wave32) -----------
// A 16x32: lane(half,m): VGPR0-7 = K 16h..16h+15 (matches B)
__device__ __forceinline__ v16bf rs_ld_frag_a(const __bf16* rowp, int half) {
    union { v16bf v; v8bf h[2]; } u;
    u.h[0] = *(const v8bf*)(rowp + 16 * half);
    u.h[1] = *(const v8bf*)(rowp + 16 * half + 8);
    return u.v;
}
// B 32x16: lane(half,n): VGPR0-7 = K 16h..16h+15 (32 contiguous bytes)
__device__ __forceinline__ v16bf rs_ld_frag_b(const __bf16* rowp, int half) {
    union { v16bf v; v8bf h[2]; } u;
    u.h[0] = *(const v8bf*)(rowp + 16 * half);
    u.h[1] = *(const v8bf*)(rowp + 16 * half + 8);
    return u.v;
}

// ---------------------------------------------------------------------------
// Pass 2: tiled NT-GEMM, bf16x3 via V_WMMA_F32_16X16X32_BF16.
// 256 threads = 8 wave32 waves (4 in M x 2 in N); wave tile 32x64 (2x4 WMMA).
// Double-buffered LDS, one barrier per K slab.
// ---------------------------------------------------------------------------
__global__ __launch_bounds__(256)
void rs_cosgemm_kernel(const float* __restrict__ A,     // [M, 1024] test reps
                       const float* __restrict__ B,     // [N, 1024] train reps
                       const float* __restrict__ invA,  // [M]
                       const float* __restrict__ invB,  // [N]
                       float* __restrict__ C,           // [M, N]
                       int N) {
    __shared__ __bf16 As_hi[2][BM][LDPB];
    __shared__ __bf16 As_lo[2][BM][LDPB];
    __shared__ __bf16 Bs_hi[2][BN][LDPB];
    __shared__ __bf16 Bs_lo[2][BN][LDPB];   // 8 * 128*40*2B = 80 KB

    const int tid  = threadIdx.x;
    const int lane = tid & 31;
    const int w    = tid >> 5;   // wave id 0..7
    const int wm   = w & 3;      // wave M coord (0..3) -> 32-row strip
    const int wn   = w >> 2;     // wave N coord (0..1) -> 64-col strip
    const int half = lane >> 4;  // 0: lanes 0-15, 1: lanes 16-31
    const int l16  = lane & 15;

    const int mBase = blockIdx.y * BM;
    const int nBase = blockIdx.x * BN;

    // Staging map: thread covers (row = idx>>3, float4-col = idx&7), idx = tid+256j
    const int sr = tid >> 3;         // staging row for j=0
    const int sc = tid & 7;          // staging float4 column

    v8f acc[2][4];
    #pragma unroll
    for (int mt = 0; mt < 2; ++mt)
        #pragma unroll
        for (int nt = 0; nt < 4; ++nt)
            acc[mt][nt] = (v8f){};

    float4 ra[4], rb[4];

    // --- prologue: fetch slab 0 and store into buffer 0
    #pragma unroll
    for (int j = 0; j < 4; ++j) {
        const int r = sr + 32 * j;   // (tid + 256j) >> 3
        ra[j] = *(const float4*)(A + (size_t)(mBase + r) * D_DIM + sc * 4);
        rb[j] = *(const float4*)(B + (size_t)(nBase + r) * D_DIM + sc * 4);
    }
    #pragma unroll
    for (int j = 0; j < 4; ++j) {
        const int r = sr + 32 * j;
        bf16x4 hi, lo;
        rs_split4(ra[j], hi, lo);
        *(bf16x4*)&As_hi[0][r][sc * 4] = hi;
        *(bf16x4*)&As_lo[0][r][sc * 4] = lo;
        rs_split4(rb[j], hi, lo);
        *(bf16x4*)&Bs_hi[0][r][sc * 4] = hi;
        *(bf16x4*)&Bs_lo[0][r][sc * 4] = lo;
    }
    __syncthreads();

    int cur = 0;
    for (int k0 = 0; k0 < D_DIM; k0 += KS) {
        const bool has_next = (k0 + KS) < D_DIM;

        // --- issue global loads for next slab (latency hides under WMMAs)
        if (has_next) {
            #pragma unroll
            for (int j = 0; j < 4; ++j) {
                const int r = sr + 32 * j;
                ra[j] = *(const float4*)(A + (size_t)(mBase + r) * D_DIM + (k0 + KS) + sc * 4);
                rb[j] = *(const float4*)(B + (size_t)(nBase + r) * D_DIM + (k0 + KS) + sc * 4);
            }
        }

        // --- compute: one 32-K WMMA step, 2x4 tiles, 3 split-terms each
        v16bf ahi[2], alo[2];
        #pragma unroll
        for (int mt = 0; mt < 2; ++mt) {
            const int rowA = wm * 32 + mt * 16 + l16;
            ahi[mt] = rs_ld_frag_a(&As_hi[cur][rowA][0], half);
            alo[mt] = rs_ld_frag_a(&As_lo[cur][rowA][0], half);
        }
        #pragma unroll
        for (int nt = 0; nt < 4; ++nt) {
            const int rowB = wn * 64 + nt * 16 + l16;
            const v16bf bhi = rs_ld_frag_b(&Bs_hi[cur][rowB][0], half);
            const v16bf blo = rs_ld_frag_b(&Bs_lo[cur][rowB][0], half);
            #pragma unroll
            for (int mt = 0; mt < 2; ++mt) {
                acc[mt][nt] = __builtin_amdgcn_wmma_f32_16x16x32_bf16(
                    false, ahi[mt], false, bhi, (short)0, acc[mt][nt], false, false);
                acc[mt][nt] = __builtin_amdgcn_wmma_f32_16x16x32_bf16(
                    false, ahi[mt], false, blo, (short)0, acc[mt][nt], false, false);
                acc[mt][nt] = __builtin_amdgcn_wmma_f32_16x16x32_bf16(
                    false, alo[mt], false, bhi, (short)0, acc[mt][nt], false, false);
            }
        }

        // --- convert + store next slab into the other buffer
        if (has_next) {
            const int nxt = cur ^ 1;
            #pragma unroll
            for (int j = 0; j < 4; ++j) {
                const int r = sr + 32 * j;
                bf16x4 hi, lo;
                rs_split4(ra[j], hi, lo);
                *(bf16x4*)&As_hi[nxt][r][sc * 4] = hi;
                *(bf16x4*)&As_lo[nxt][r][sc * 4] = lo;
                rs_split4(rb[j], hi, lo);
                *(bf16x4*)&Bs_hi[nxt][r][sc * 4] = hi;
                *(bf16x4*)&Bs_lo[nxt][r][sc * 4] = lo;
            }
        }
        __syncthreads();
        cur ^= 1;
    }

    // --- epilogue: C/D layout VGPR r -> (M = r + 8*half, N = l16) per tile
    const int rowBase = mBase + wm * 32;
    const int colBase = nBase + wn * 64;
    #pragma unroll
    for (int mt = 0; mt < 2; ++mt) {
        float ia[8];
        #pragma unroll
        for (int r = 0; r < 8; ++r)
            ia[r] = invA[rowBase + mt * 16 + half * 8 + r];
        #pragma unroll
        for (int nt = 0; nt < 4; ++nt) {
            const int col = colBase + nt * 16 + l16;
            const float ib = invB[col];
            #pragma unroll
            for (int r = 0; r < 8; ++r) {
                const int row = rowBase + mt * 16 + half * 8 + r;
                C[(size_t)row * N + col] = acc[mt][nt][r] * ia[r] * ib;
            }
        }
    }
}

// ---------------------------------------------------------------------------
// Host launcher
// ---------------------------------------------------------------------------
extern "C" void kernel_launch(void* const* d_in, const int* in_sizes, int n_in,
                              void* d_out, int out_size, void* d_ws, size_t ws_size,
                              hipStream_t stream) {
    const float* H = (const float*)d_in[0];       // [Nx, 1024]
    const float* T = (const float*)d_in[1];       // [Nt, 1024]
    float* out     = (float*)d_out;               // [Nx, Nt]

    const int Nx = in_sizes[0] / D_DIM;           // 2048
    const int Nt = in_sizes[1] / D_DIM;           // 16384

    float* invA = (float*)d_ws;                   // [Nx]
    float* invB = invA + Nx;                      // [Nt]  (total 73.7 KB of ws)

    rs_rownorm_kernel<<<Nx, 256, 0, stream>>>(H, invA);
    rs_rownorm_kernel<<<Nt, 256, 0, stream>>>(T, invB);

    dim3 grid(Nt / BN, Nx / BM);                  // 128 x 16 workgroups
    rs_cosgemm_kernel<<<grid, 256, 0, stream>>>(H, T, invA, invB, out, Nt);
}
